// DereverbLoss_21835613733270
// MI455X (gfx1250) — compile-verified, matrix operand order checked by reading, not run on hardware
//
#include <hip/hip_runtime.h>
#include <hip/hip_bf16.h>

// ---------------------------------------------------------------------------
// CDNA5 (gfx1250) dereverb loss. wave32, WMMA bf16 for the mel GEMM with
// b128 fragment loads, LDS FFTs for STFT + partitioned convolution,
// global f32 atomics for scalar reductions.
// ---------------------------------------------------------------------------

typedef __attribute__((ext_vector_type(16))) __bf16 v16bf;
typedef __attribute__((ext_vector_type(8)))  float  v8f;

#define T_LEN   160000
#define NSIG    32           // B*C = 16*2
#define NFRAMES 626
#define NROWS   (NSIG * NFRAMES)   // 20032, multiple of 16
#define KPAD    544                // 513 freq bins padded to multiple of 32
#define NMELS   80
#define NBLK    157                // ceil(160000/1024) partitions for conv
#define WSUM_F  384.0f             // sum(hann^2) for N=1024 exactly
#define INV_WSQRT 0.05103103630798288f  // 1/sqrt(384)

// ---------------- reduction helpers (wave32) ----------------
__device__ __forceinline__ float warp_sum32(float v) {
#pragma unroll
  for (int o = 16; o > 0; o >>= 1) v += __shfl_down(v, o, 32);
  return v;
}

__device__ __forceinline__ void block_atomic_sum(float v, float* dst, float* sbuf) {
  v = warp_sum32(v);
  const int lane = threadIdx.x & 31;
  const int wid  = threadIdx.x >> 5;
  if (lane == 0) sbuf[wid] = v;
  __syncthreads();
  if (threadIdx.x == 0) {
    float s = 0.f;
    const int nw = (blockDim.x + 31) >> 5;
    for (int i = 0; i < nw; ++i) s += sbuf[i];
    atomicAdd(dst, s);
  }
  __syncthreads();
}

// ---------------- in-LDS iterative radix-2 FFT ----------------
// dir = -1 : forward (e^{-i2pi kn/N}), dir = +1 : inverse (unscaled)
template <int LOGN>
__device__ void fft_inplace(float* re, float* im, float dir) {
  const int N = 1 << LOGN;
  const int tid = threadIdx.x;
  const int nt  = blockDim.x;
  // bit-reverse permutation
  for (int i = tid; i < N; i += nt) {
    unsigned j = __brev((unsigned)i) >> (32 - LOGN);
    if (j > (unsigned)i) {
      float tr = re[i]; re[i] = re[j]; re[j] = tr;
      float ti = im[i]; im[i] = im[j]; im[j] = ti;
    }
  }
  __syncthreads();
  for (int s = 1; s <= LOGN; ++s) {
    const int m  = 1 << s;
    const int mh = m >> 1;
    for (int b = tid; b < (N >> 1); b += nt) {
      const int grp = b >> (s - 1);
      const int pos = b & (mh - 1);
      const int i0 = grp * m + pos;
      const int i1 = i0 + mh;
      const float ang = dir * 6.283185307179586f * (float)pos / (float)m;
      float wi, wr;
      sincosf(ang, &wi, &wr);
      const float xr = re[i1], xi = im[i1];
      const float tr = wr * xr - wi * xi;
      const float ti = wr * xi + wi * xr;
      const float ur = re[i0], ui = im[i0];
      re[i0] = ur + tr; im[i0] = ui + ti;
      re[i1] = ur - tr; im[i1] = ui - ti;
    }
    __syncthreads();
  }
}

// ---------------- kernel: zero f32 buffer ----------------
__global__ void zero_f32_kernel(float* __restrict__ p, size_t n) {
  for (size_t i = blockIdx.x * (size_t)blockDim.x + threadIdx.x; i < n;
       i += (size_t)gridDim.x * blockDim.x)
    p[i] = 0.f;
}

// ---------- kernel: build TRANSPOSED mel filterbank (bf16, [80][KPAD]) -----
// Transposed so a WMMA B-fragment column is K-contiguous (b128 loads).
__global__ void fbank_kernel(__bf16* __restrict__ FBt) {
  const int idx = blockIdx.x * blockDim.x + threadIdx.x;
  if (idx >= NMELS * KPAD) return;
  const int m = idx / KPAD;
  const int k = idx % KPAD;
  float val = 0.f;
  if (k < 513) {
    const float f = 8000.0f * (float)k / 512.0f;
    const float mmax = 2595.0f * log10f(1.0f + 8000.0f / 700.0f);
    auto m2h = [](float mv) { return 700.0f * (powf(10.0f, mv / 2595.0f) - 1.0f); };
    const float p0 = m2h(mmax * (float)(m)     / 81.0f);
    const float p1 = m2h(mmax * (float)(m + 1) / 81.0f);
    const float p2 = m2h(mmax * (float)(m + 2) / 81.0f);
    const float down = -(p0 - f) / (p1 - p0);
    const float up   =  (p2 - f) / (p2 - p1);
    val = fmaxf(0.f, fminf(down, up));
  }
  FBt[(size_t)m * KPAD + k] = (__bf16)val;
}

// ---------------- kernel: time-domain L1 + rir regularizer pieces ----------
// acc[0]=dryL1 acc[1]=rirL1 acc[2]=sparsity acc[3..6]=segment energies
__global__ void time_reduce_kernel(const float* __restrict__ est,
                                   const float* __restrict__ src,
                                   float* __restrict__ acc) {
  __shared__ float sbuf[8];
  const size_t N = (size_t)NSIG * T_LEN;  // 5,120,000 per plane
  float dsum = 0.f, rsum = 0.f, spar = 0.f;
  float e0 = 0.f, e1 = 0.f, e2 = 0.f, e3 = 0.f;
  const size_t stride = (size_t)gridDim.x * blockDim.x;
  for (size_t i = blockIdx.x * (size_t)blockDim.x + threadIdx.x; i < N; i += stride) {
    const size_t bc = i / T_LEN;
    const size_t t  = i % T_LEN;
    const size_t b  = bc >> 1, c = bc & 1;
    const size_t dryIdx = ((b * 2 + 0) * 2 + c) * (size_t)T_LEN + t;
    const size_t rirIdx = ((b * 2 + 1) * 2 + c) * (size_t)T_LEN + t;
    __builtin_prefetch(&est[dryIdx + stride], 0, 1);   // global_prefetch_b8
    __builtin_prefetch(&src[dryIdx + stride], 0, 1);
    const float pd = est[dryIdx], td = src[dryIdx];
    const float pr = est[rirIdx], tr = src[rirIdx];
    dsum += fabsf(pd - td);
    rsum += fabsf(pr - tr);
    spar += fabsf(pr);
    const float v2 = pr * pr;
    const int seg = (int)(t / 40000);
    e0 += (seg == 0) ? v2 : 0.f;
    e1 += (seg == 1) ? v2 : 0.f;
    e2 += (seg == 2) ? v2 : 0.f;
    e3 += (seg == 3) ? v2 : 0.f;
  }
  block_atomic_sum(dsum, &acc[0], sbuf);
  block_atomic_sum(rsum, &acc[1], sbuf);
  block_atomic_sum(spar, &acc[2], sbuf);
  block_atomic_sum(e0,   &acc[3], sbuf);
  block_atomic_sum(e1,   &acc[4], sbuf);
  block_atomic_sum(e2,   &acc[5], sbuf);
  block_atomic_sum(e3,   &acc[6], sbuf);
}

// ---------------- kernel: STFT (pred packed in re, target in im) ----------
// acc[7]=sum |(|P|-|T|)|/sqrt(WSUM), acc[8]=sum cos(dphi)
// Pp/Tp: bf16 power spectra rows [NROWS][KPAD] for the WMMA mel GEMM.
__global__ void stft_loss_kernel(const float* __restrict__ est,
                                 const float* __restrict__ src,
                                 __bf16* __restrict__ Pp,
                                 __bf16* __restrict__ Tp,
                                 float* __restrict__ acc) {
  __shared__ float re[1024], im[1024];
  __shared__ float sbuf[8];
  const int frame = blockIdx.x;    // 0..625
  const int sig   = blockIdx.y;    // 0..31
  const int b = sig >> 1, c = sig & 1;
  const float* pd = est + ((size_t)(b * 2 + 0) * 2 + c) * T_LEN;  // pred dry
  const float* td = src + ((size_t)(b * 2 + 0) * 2 + c) * T_LEN;  // target dry
  const int tid = threadIdx.x;

  for (int n = tid; n < 1024; n += blockDim.x) {
    int j = frame * 256 + n - 512;                 // reflect pad 512
    if (j < 0) j = -j;
    if (j >= T_LEN) j = 2 * T_LEN - 2 - j;
    const float w = 0.5f - 0.5f * cosf(6.283185307179586f * (float)n / 1024.0f);
    re[n] = pd[j] * w;
    im[n] = td[j] * w;
  }
  __syncthreads();
  fft_inplace<10>(re, im, -1.0f);

  float magsum = 0.f, cossum = 0.f;
  const size_t row = (size_t)sig * NFRAMES + frame;
  for (int k = tid; k < KPAD; k += blockDim.x) {
    if (k <= 512) {
      const int q = (1024 - k) & 1023;
      // Hermitian unpack: Z = P + iT, both p,t real
      const float Pr = 0.5f * (re[k] + re[q]);
      const float Pi = 0.5f * (im[k] - im[q]);
      const float Tr = 0.5f * (im[k] + im[q]);
      const float Ti = 0.5f * (re[q] - re[k]);
      const float pp = Pr * Pr + Pi * Pi;
      const float tp = Tr * Tr + Ti * Ti;
      const float magp = sqrtf(pp), magt = sqrtf(tp);
      magsum += fabsf(magp - magt) * INV_WSQRT;
      const float denom = magp * magt;
      cossum += (denom > 1e-30f) ? (Pr * Tr + Pi * Ti) / denom : 1.0f;
      Pp[row * KPAD + k] = (__bf16)(pp * (1.0f / WSUM_F));
      Tp[row * KPAD + k] = (__bf16)(tp * (1.0f / WSUM_F));
    } else {  // K padding must be clean zeros for WMMA (0*Inf -> NaN)
      Pp[row * KPAD + k] = (__bf16)0.f;
      Tp[row * KPAD + k] = (__bf16)0.f;
    }
  }
  block_atomic_sum(magsum, &acc[7], sbuf);
  block_atomic_sum(cossum, &acc[8], sbuf);
}

// ---------------- kernel: mel GEMM via v_wmma_f32_16x16x32_bf16 -----------
// 5 waves per block, wave w handles mel tile w (80 = 5 * 16).
// CDNA5 16-bit A layout is K-contiguous per lane half: VGPR0..3 hold
// K=khalf..khalf+7 (16 bytes), VGPR4..7 hold K=16+khalf..+7 (16 bytes),
// so each fragment is exactly two b128 loads.
// acc[9] = sum |log(pm+eps) - log(tm+eps)|
union Frag128 {
  uint4 q[2];
  v16bf v;
};

__device__ __forceinline__ v16bf load_frag(const __bf16* __restrict__ base,
                                           int k0, int khalf) {
  Frag128 f;
  f.q[0] = *reinterpret_cast<const uint4*>(base + k0 + khalf);        // K khalf..+7
  f.q[1] = *reinterpret_cast<const uint4*>(base + k0 + 16 + khalf);   // K 16+khalf..+7
  return f.v;
}

__global__ void mel_wmma_kernel(const __bf16* __restrict__ Pp,
                                const __bf16* __restrict__ Tp,
                                const __bf16* __restrict__ FBt,
                                float* __restrict__ acc) {
  __shared__ float sbuf[8];
  const int rowBase = blockIdx.x * 16;
  const int wave    = threadIdx.x >> 5;    // 0..4 -> mel tile
  const int lane    = threadIdx.x & 31;
  const int melBase = wave * 16;
  const int mrow    = lane & 15;
  const int khalf   = (lane >> 4) * 8;     // 16-bit layout half-wave K offset
  const __bf16* prow = Pp  + (size_t)(rowBase + mrow) * KPAD;
  const __bf16* trow = Tp  + (size_t)(rowBase + mrow) * KPAD;
  const __bf16* frow = FBt + (size_t)(melBase + mrow) * KPAD;  // B col, K-contig

  v8f accP = {};
  v8f accT = {};
#pragma unroll 2
  for (int k0 = 0; k0 < KPAD; k0 += 32) {
    const v16bf a_p = load_frag(prow, k0, khalf);
    const v16bf a_t = load_frag(trow, k0, khalf);
    const v16bf bm  = load_frag(frow, k0, khalf);
    accP = __builtin_amdgcn_wmma_f32_16x16x32_bf16(false, a_p, false, bm,
                                                   (short)0, accP, false, false);
    accT = __builtin_amdgcn_wmma_f32_16x16x32_bf16(false, a_t, false, bm,
                                                   (short)0, accT, false, false);
  }
  // C/D layout: VGPR r -> M = r (+8 for upper lanes); all rows/mels valid.
  float s = 0.f;
#pragma unroll
  for (int r = 0; r < 8; ++r)
    s += fabsf(logf(accP[r] + 1e-8f) - logf(accT[r] + 1e-8f));
  block_atomic_sum(s, &acc[9], sbuf);
}

// ---------------- kernel: 2048-pt partition FFTs for consistency ----------
// dry packed in re, rir in im -> one FFT yields both spectra (1025 bins kept).
__global__ void fft_blocks_kernel(const float* __restrict__ est,
                                  float2* __restrict__ DryF,
                                  float2* __restrict__ RirF) {
  __shared__ float re[2048], im[2048];
  const int p   = blockIdx.x;   // partition 0..156
  const int sig = blockIdx.y;   // 0..31
  const int b = sig >> 1, c = sig & 1;
  const float* dry = est + ((size_t)(b * 2 + 0) * 2 + c) * T_LEN;
  const float* rir = est + ((size_t)(b * 2 + 1) * 2 + c) * T_LEN;
  const int tid = threadIdx.x;
  for (int n = tid; n < 2048; n += blockDim.x) {
    const int g = p * 1024 + n;
    const bool valid = (n < 1024) && (g < T_LEN);
    re[n] = valid ? dry[g] : 0.f;
    im[n] = valid ? rir[g] : 0.f;
  }
  __syncthreads();
  fft_inplace<11>(re, im, -1.0f);
  const size_t base = ((size_t)sig * NBLK + p) * 1025;
  for (int k = tid; k <= 1024; k += blockDim.x) {
    const int q = (2048 - k) & 2047;
    DryF[base + k] = make_float2(0.5f * (re[k] + re[q]), 0.5f * (im[k] - im[q]));
    RirF[base + k] = make_float2(0.5f * (im[k] + im[q]), 0.5f * (re[q] - re[k]));
  }
}

// ---------------- kernel: spectral pair-sum + IFFT + overlap-add ----------
__global__ void conv_ifft_kernel(const float2* __restrict__ DryF,
                                 const float2* __restrict__ RirF,
                                 float* __restrict__ Conv) {
  __shared__ float re[2048], im[2048];
  __shared__ float sr[1025], si[1025];
  const int o   = blockIdx.x;   // output block 0..156
  const int sig = blockIdx.y;
  const int tid = threadIdx.x;
  for (int k = tid; k < 1025; k += blockDim.x) { sr[k] = 0.f; si[k] = 0.f; }
  __syncthreads();
  const size_t sb = (size_t)sig * NBLK;
  for (int j = 0; j <= o; ++j) {   // S_o = sum_{i+j=o} D_i * R_j
    const float2* D = DryF + (sb + (size_t)(o - j)) * 1025;
    const float2* R = RirF + (sb + (size_t)j) * 1025;
    for (int k = tid; k < 1025; k += blockDim.x) {   // each thread owns fixed k
      const float2 d = D[k], r = R[k];
      sr[k] += d.x * r.x - d.y * r.y;
      si[k] += d.x * r.y + d.y * r.x;
    }
  }
  __syncthreads();
  for (int k = tid; k < 2048; k += blockDim.x) {     // Hermitian reconstruct
    if (k <= 1024) { re[k] = sr[k];        im[k] = si[k]; }
    else           { re[k] = sr[2048 - k]; im[k] = -si[2048 - k]; }
  }
  __syncthreads();
  fft_inplace<11>(re, im, +1.0f);
  for (int n = tid; n < 2048; n += blockDim.x) {
    const int g = o * 1024 + n;
    if (g < T_LEN)
      atomicAdd(&Conv[(size_t)sig * T_LEN + g], re[n] * (1.0f / 2048.0f));
  }
}

// ---------------- kernel: consistency L1 vs mix ----------
__global__ void consistency_l1_kernel(const float* __restrict__ Conv,
                                      const float* __restrict__ mix,
                                      float* __restrict__ acc) {
  __shared__ float sbuf[8];
  const size_t N = (size_t)NSIG * T_LEN;
  float s = 0.f;
  for (size_t i = blockIdx.x * (size_t)blockDim.x + threadIdx.x; i < N;
       i += (size_t)gridDim.x * blockDim.x) {
    __builtin_prefetch(&mix[i + (size_t)gridDim.x * blockDim.x], 0, 1);
    s += fabsf(Conv[i] - mix[i]);
  }
  block_atomic_sum(s, &acc[10], sbuf);
}

// ---------------- kernel: combine everything ----------
__global__ void finalize_kernel(const float* __restrict__ acc, float* __restrict__ out) {
  if (threadIdx.x != 0 || blockIdx.x != 0) return;
  const float NT   = 5120000.0f;     // 32*160000
  const float NS   = 10276416.0f;    // 32*626*513
  const float NM   = 1602560.0f;     // 32*626*80
  const float NSEG = 1280000.0f;     // 32*40000
  const float dry_time   = acc[0] / NT;
  const float mag_loss   = acc[7] / NS;
  const float phase_loss = 1.0f - acc[8] / NS;
  const float dry_freq   = mag_loss + 0.1f * phase_loss;
  const float dry_mel    = acc[9] / NM;
  const float total_dry  = dry_time + 0.5f * dry_freq + 0.3f * dry_mel;
  const float rir_time = acc[1] / NT;
  const float spars    = acc[2] / NT;
  const float e0 = acc[3] / NSEG, e1 = acc[4] / NSEG;
  const float e2 = acc[5] / NSEG, e3 = acc[6] / NSEG;
  const float decay = fmaxf(0.f, e1 - 0.8f * e0) + fmaxf(0.f, e2 - 0.8f * e1) +
                      fmaxf(0.f, e3 - 0.8f * e2);
  const float total_rir = rir_time + 0.1f * (spars + decay);
  const float consistency = acc[10] / NT;
  out[0] = 3.0f * total_dry + 1.0f * total_rir + 0.2f * consistency;
}

// ---------------------------------------------------------------------------
extern "C" void kernel_launch(void* const* d_in, const int* in_sizes, int n_in,
                              void* d_out, int out_size, void* d_ws, size_t ws_size,
                              hipStream_t stream) {
  const float* est = (const float*)d_in[0];
  const float* src = (const float*)d_in[1];
  const float* mix = (const float*)d_in[2];
  float* out = (float*)d_out;

  // workspace layout
  char* ws = (char*)d_ws;
  auto alignup = [](size_t x) { return (x + 255) & ~(size_t)255; };
  float* acc = (float*)ws;
  size_t off = 256;
  __bf16* Pp = (__bf16*)(ws + off); off = alignup(off + (size_t)NROWS * KPAD * sizeof(__bf16));
  __bf16* Tp = (__bf16*)(ws + off); off = alignup(off + (size_t)NROWS * KPAD * sizeof(__bf16));
  __bf16* FBt = (__bf16*)(ws + off); off = alignup(off + (size_t)NMELS * KPAD * sizeof(__bf16));
  float2* DryF = (float2*)(ws + off); off = alignup(off + (size_t)NSIG * NBLK * 1025 * sizeof(float2));
  float2* RirF = (float2*)(ws + off); off = alignup(off + (size_t)NSIG * NBLK * 1025 * sizeof(float2));
  float*  Conv = (float*)(ws + off);  off = alignup(off + (size_t)NSIG * T_LEN * sizeof(float));
  (void)ws_size; (void)in_sizes; (void)n_in; (void)out_size;

  zero_f32_kernel<<<1, 64, 0, stream>>>(acc, 16);
  zero_f32_kernel<<<2048, 256, 0, stream>>>(Conv, (size_t)NSIG * T_LEN);
  fbank_kernel<<<(NMELS * KPAD + 255) / 256, 256, 0, stream>>>(FBt);

  time_reduce_kernel<<<2048, 256, 0, stream>>>(est, src, acc);

  dim3 gs(NFRAMES, NSIG);
  stft_loss_kernel<<<gs, 256, 0, stream>>>(est, src, Pp, Tp, acc);
  mel_wmma_kernel<<<NROWS / 16, 160, 0, stream>>>(Pp, Tp, FBt, acc);

  dim3 gf(NBLK, NSIG);
  fft_blocks_kernel<<<gf, 256, 0, stream>>>(est, DryF, RirF);
  conv_ifft_kernel<<<gf, 256, 0, stream>>>(DryF, RirF, Conv);
  consistency_l1_kernel<<<2048, 256, 0, stream>>>(Conv, mix, acc);

  finalize_kernel<<<1, 32, 0, stream>>>(acc, out);
}
